// MOEDIVAESR_14164802142766
// MI455X (gfx1250) — matrix operand order repaired and checked by reading, not compiled, not run on hardware
//
#include <hip/hip_runtime.h>
#include <hip/hip_bf16.h>

typedef __attribute__((ext_vector_type(16))) _Float16 v16h;
typedef __attribute__((ext_vector_type(8)))  _Float16 h8;
typedef __attribute__((ext_vector_type(8)))  float    v8f;
typedef __attribute__((ext_vector_type(4)))  unsigned int u32x4;
typedef __attribute__((ext_vector_type(4)))  int i32x4;
typedef __attribute__((ext_vector_type(8)))  int i32x8;

#define NB    64
#define IMG   256
#define GFEAT 64
#define NCH   6
#define NPA   2
#define CMID  8

// ---- A-fragment fills, specialized on half-wave (HI8) and K-chunk so all
// ---- k/ky/kx become compile-time constants (ISA 7.12.2 16-bit A 16x32:
// ---- elems 0..7 -> K = HI8+j, elems 8..15 -> K = 16+HI8+(j-8), M = lane%16; a_gate/a_c1 use K = 2*HI8+j).

// gating conv: K = ky*7+kx (k < 49), f16 LDS
template <int HI8, int CK>
__device__ __forceinline__ v16h a_gate(const _Float16 (*xs)[260], int orow, int col) {
  v16h a;
#pragma unroll
  for (int j = 0; j < 16; ++j) {
    const int k = CK * 32 + 2 * HI8 + j;
    a[j] = (k < 49) ? xs[orow * 4 + k / 7][col * 4 + k % 7] : (_Float16)0.f;
  }
  return a;
}

// expert conv1: K = ky*3+kx (k < 9), f32 LDS (TDM-staged), cvt on read
template <int HI8>
__device__ __forceinline__ v16h a_c1(const float (*xs)[260], int hrow, int col) {
  v16h a;
#pragma unroll
  for (int j = 0; j < 16; ++j) {
    const int k = 2 * HI8 + j;
    a[j] = (k < 9) ? (_Float16)xs[hrow + k / 3][col + k % 3] : (_Float16)0.f;
  }
  return a;
}

// expert conv2: tap-major K = tap*8 + c -> each 8-elem half of the fragment is
// 8 contiguous aligned channels => one ds_load_b128 per half.
template <int HI8, int CK>
__device__ __forceinline__ v16h a_c2(const _Float16 (*hs)[260][CMID], int hl, int col) {
  const int t0 = CK * 4 + (HI8 ? 1 : 0);
  const int t1 = CK * 4 + (HI8 ? 3 : 2);
  h8 z = {};
  h8 lo = (t0 <= 8) ? *(const h8*)&hs[hl + t0 / 3 - 1][col + t0 % 3][0] : z;
  h8 hi = (t1 <= 8) ? *(const h8*)&hs[hl + t1 / 3 - 1][col + t1 % 3][0] : z;
  v16h a;
#pragma unroll
  for (int j = 0; j < 8; ++j) { a[j] = lo[j]; a[j + 8] = hi[j]; }
  return a;
}

// ---------------------------------------------------------------- kernel 0
__global__ void zero_feats(float* feats) {
  int i = blockIdx.x * blockDim.x + threadIdx.x;
  if (i < NB * GFEAT) feats[i] = 0.f;
}

// ---------------------------------------------------------------- kernel 1
// Gating conv 7x7 stride4 (1->64) + relu + spatial mean, as WMMA GEMM.
// grid = (4 bands, 64 samples), 256 threads (8 wave32).
__global__ __launch_bounds__(256) void gating_conv_wmma(
    const float* __restrict__ x, const float* __restrict__ Wg,
    float* __restrict__ feats) {
  __shared__ _Float16 xs[68][260];
  __shared__ _Float16 bw[64][64];
  __shared__ float fsum[GFEAT];

  const int band = blockIdx.x;
  const int s    = blockIdx.y;
  const int tid  = threadIdx.x;
  const int lane = tid & 31;
  const int wave = tid >> 5;
  const int nloc = lane & 15;

  const float* xp = x + (size_t)s * (IMG * IMG);
  for (int i = tid; i < 68 * 260; i += 256) {
    int r = i / 260, c = i % 260;
    int iy = band * 64 - 1 + r;
    int ix = c - 1;
    float v = 0.f;
    if (iy >= 0 && iy < IMG && ix >= 0 && ix < IMG) {
      const float* p = xp + iy * IMG + ix;
      __builtin_prefetch(p + IMG, 0, 0);          // global_prefetch_b8
      v = *p;
    }
    xs[r][c] = (_Float16)v;
  }
  for (int i = tid; i < 64 * 64; i += 256) {
    int k = i >> 6, n = i & 63;
    bw[k][n] = (k < 49) ? (_Float16)Wg[n * 49 + k] : (_Float16)0.f;
  }
  if (tid < GFEAT) fsum[tid] = 0.f;
  __syncthreads();

  // B tiles (32x16 f16: lanes 0-15 K=0..15, lanes 16-31 K=16..31)
  const int kb = (lane < 16) ? 0 : 16;
  v16h breg[4][2];
#pragma unroll
  for (int nt = 0; nt < 4; ++nt)
#pragma unroll
    for (int ck = 0; ck < 2; ++ck)
#pragma unroll
      for (int j = 0; j < 16; ++j)
        breg[nt][ck][j] = bw[ck * 32 + kb + j][nt * 16 + nloc];

  float chsum[4] = {0.f, 0.f, 0.f, 0.f};

  for (int gi = 0; gi < 8; ++gi) {
    int g    = wave * 8 + gi;                     // uniform per wave
    int orow = g >> 2;
    int oc0  = (g & 3) << 4;
    int col  = oc0 + nloc;
    v16h a0 = (lane < 16) ? a_gate<0, 0>(xs, orow, col) : a_gate<8, 0>(xs, orow, col);
    v16h a1 = (lane < 16) ? a_gate<0, 1>(xs, orow, col) : a_gate<8, 1>(xs, orow, col);
#pragma unroll
    for (int nt = 0; nt < 4; ++nt) {
      v8f acc = {};
      acc = __builtin_amdgcn_wmma_f32_16x16x32_f16(false, a0, false, breg[nt][0],
                                                   (short)0, acc, false, false);
      acc = __builtin_amdgcn_wmma_f32_16x16x32_f16(false, a1, false, breg[nt][1],
                                                   (short)0, acc, false, false);
      float p = 0.f;
#pragma unroll
      for (int r = 0; r < 8; ++r) p += (acc[r] > 0.f) ? acc[r] : 0.f;
      chsum[nt] += p;
    }
  }
#pragma unroll
  for (int nt = 0; nt < 4; ++nt)
    atomicAdd(&fsum[nt * 16 + nloc], chsum[nt]);  // ds_add_f32
  __syncthreads();
  if (tid < GFEAT)
    atomicAdd(&feats[s * GFEAT + tid], fsum[tid] * (1.f / 4096.f));
}

// ---------------------------------------------------------------- kernel 2
__global__ void gating_head(const float* __restrict__ feats,
                            const float* __restrict__ Wc, const float* __restrict__ bc,
                            const float* __restrict__ Wp, const float* __restrict__ bp,
                            int* __restrict__ gidx, float* __restrict__ gpr) {
  int s = threadIdx.x;
  if (s >= NB) return;
  const float* f = feats + s * GFEAT;
  float lc[NCH], lp[NPA];
  for (int j = 0; j < NCH; ++j) {
    float d = bc[j];
    for (int k = 0; k < GFEAT; ++k) d += f[k] * Wc[j * GFEAT + k];
    lc[j] = d;                                    // T == 1.0
  }
  for (int j = 0; j < NPA; ++j) {
    float d = bp[j];
    for (int k = 0; k < GFEAT; ++k) d += f[k] * Wp[j * GFEAT + k];
    lp[j] = d;
  }
  int ci = 0; float cm = lc[0];
  for (int j = 1; j < NCH; ++j) if (lc[j] > cm) { cm = lc[j]; ci = j; }
  float cs = 0.f;
  for (int j = 0; j < NCH; ++j) cs += __expf(lc[j] - cm);
  float cval = 1.f / cs;
  int pi = 0; float pm = lp[0];
  for (int j = 1; j < NPA; ++j) if (lp[j] > pm) { pm = lp[j]; pi = j; }
  float ps = 0.f;
  for (int j = 0; j < NPA; ++j) ps += __expf(lp[j] - pm);
  float pval = 1.f / ps;
  float wp = 0.3f * pval, wc = 0.7f * cval;
  float tot = wp + wc + 1e-8f;
  gidx[s * 2 + 0] = pi;
  gidx[s * 2 + 1] = ci + 2;
  gpr[s * 2 + 0] = wp / tot;
  gpr[s * 2 + 1] = wc / tot;
}

// ---------------------------------------------------------------- kernel 3
// Fused per-sample 2-expert conv1(1->8)+relu -> conv2(8->1), weighted combine.
// x band staged HBM->LDS by the Tensor Data Mover. grid = (32, 64), 256 thr.
__global__ __launch_bounds__(256) void expert_moe_wmma(
    const float* __restrict__ x,
    const float* __restrict__ We1, const float* __restrict__ be1,
    const float* __restrict__ We2, const float* __restrict__ be2,
    const int* __restrict__ gidx, const float* __restrict__ gpr,
    float* __restrict__ out) {
  __shared__ float xs32[12][260];                               // 12.5 KB
  __shared__ __attribute__((aligned(16))) _Float16 hs[10][260][CMID]; // 41.6 KB
  __shared__ float oacc[8][256];                                // 8 KB

  const int band = blockIdx.x;
  const int s    = blockIdx.y;
  const int tid  = threadIdx.x;
  const int lane = tid & 31;
  const int wave = tid >> 5;
  const int hi8  = (lane >= 16) ? 8 : 0;
  const int nloc = lane & 15;
  const int r0   = band * 8;

  const float* xp = x + (size_t)s * (IMG * IMG);

  // valid global rows [gy0, gy1] of the 12-row halo band [r0-2, r0+9]
  const int gy0  = (r0 - 2 < 0) ? 0 : (r0 - 2);
  const int gy1  = (r0 + 9 > IMG - 1) ? (IMG - 1) : (r0 + 9);
  const int rows = gy1 - gy0 + 1;
  const int l0   = gy0 - (r0 - 2);       // first LDS row the TDM writes

#if __has_builtin(__builtin_amdgcn_tensor_load_to_lds)
  // zero-fill only cells the TDM will NOT write (disjoint => no race)
  for (int i = tid; i < 12 * 260; i += 256) {
    int r = i / 260, c = i % 260;
    bool written = (r >= l0) && (r < l0 + rows) && (c >= 1) && (c <= 256);
    if (!written) xs32[r][c] = 0.f;
  }
  if (tid < 32) {                        // one wave issues the DMA
    unsigned long long ga = (unsigned long long)(uintptr_t)(xp + (size_t)gy0 * IMG);
    // generic LDS pointer: low 32 bits are the LDS byte offset (aperture form)
    unsigned ldsoff = (unsigned)(uintptr_t)&xs32[l0][1];
    u32x4 g0;
    g0[0] = 1u;                                   // count = 1 valid descriptor
    g0[1] = ldsoff;                               // lds_addr
    g0[2] = (unsigned)ga;                         // global_addr[31:0]
    g0[3] = (unsigned)(ga >> 32) | 0x80000000u;   // global_addr[56:32] | type=2
    i32x8 g1;
    g1[0] = (int)((2u << 16)                      // data_size = 4 bytes
                | (1u << 20)                      // pad_enable
                | (7u << 22)                      // pad_interval: 256 DWORDs/row
                | (3u << 25));                    // pad_amount: 4 DWORDs -> pitch 260
    g1[1] = (int)(((unsigned)IMG & 0xFFFFu) << 16);          // tensor_dim0 lo16
    g1[2] = (int)(((unsigned)IMG >> 16) | ((unsigned)rows << 16)); // dim0 hi | dim1 lo
    g1[3] = (int)(((unsigned)rows >> 16) | ((unsigned)IMG << 16)); // dim1 hi | tile_dim0
    g1[4] = rows;                                 // tile_dim1 (tile_dim2 = 0)
    g1[5] = IMG;                                  // tensor_dim0_stride
    g1[6] = 0;
    g1[7] = 0;
    i32x4 gz = {};
#if defined(__clang_major__) && (__clang_major__ >= 23)
    i32x8 gz8 = {};
    __builtin_amdgcn_tensor_load_to_lds(g0, g1, gz, gz, gz8, 0);
#else
    __builtin_amdgcn_tensor_load_to_lds(g0, g1, gz, gz, 0);
#endif
    __builtin_amdgcn_s_wait_tensorcnt(0);
  }
#else
  for (int i = tid; i < 12 * 260; i += 256) {
    int r = i / 260, c = i % 260;
    int iy = r0 - 2 + r, ix = c - 1;
    xs32[r][c] = (iy >= 0 && iy < IMG && ix >= 0 && ix < IMG) ? xp[iy * IMG + ix] : 0.f;
  }
#endif
  for (int i = tid; i < 10 * 260 * CMID; i += 256)
    ((_Float16*)hs)[i] = (_Float16)0.f;           // halos stay zero (SAME padding)
  for (int i = tid; i < 8 * 256; i += 256)
    ((float*)oacc)[i] = 0.f;
  __syncthreads();

  const int kb = (lane < 16) ? 0 : 16;

  for (int pass = 0; pass < 2; ++pass) {          // parent expert, then child
    const int   e = gidx[s * 2 + pass];
    const float w = gpr[s * 2 + pass];

    // conv1 B: B1[k][n] = We1[e][n][0][ky][kx], k = ky*3+kx < 9, n < 8
    v16h b1;
#pragma unroll
    for (int j = 0; j < 16; ++j) {
      int k = kb + j;
      b1[j] = (k < 9 && nloc < CMID) ? (_Float16)We1[e * (CMID * 9) + nloc * 9 + k]
                                     : (_Float16)0.f;
    }
    // hidden: 10 rows x 16 col-tiles = 160 tiles, 20 per wave (wave-uniform)
    for (int t = wave * 20; t < wave * 20 + 20; ++t) {
      int hrow = t >> 4;
      int c0   = (t & 15) << 4;
      int grow = r0 - 1 + hrow;
      if (grow < 0 || grow >= IMG) continue;      // stays zero-padded
      int col = c0 + nloc;
      v16h a = (lane < 16) ? a_c1<0>(xs32, hrow, col) : a_c1<8>(xs32, hrow, col);
      v8f acc = {};
      acc = __builtin_amdgcn_wmma_f32_16x16x32_f16(false, a, false, b1,
                                                   (short)0, acc, false, false);
      if (nloc < CMID) {
        float bb = be1[e * CMID + nloc];
#pragma unroll
        for (int r = 0; r < 8; ++r) {
          float h = acc[r] + bb;
          h = (h > 0.f) ? h : 0.f;
          hs[hrow][c0 + r + hi8 + 1][nloc] = (_Float16)h;
        }
      }
    }
    __syncthreads();

    // conv2 B, tap-major K: B2[k][0] = We2[e][0][c][tap], k = tap*8 + c < 72
    v16h b2[3];
#pragma unroll
    for (int ck = 0; ck < 3; ++ck)
#pragma unroll
      for (int j = 0; j < 16; ++j) {
        int k = ck * 32 + ((j < 8) ? (kb / 2 + j) : (16 + kb / 2 + (j - 8)));
        b2[ck][j] = (nloc == 0 && k < 72)
                        ? (_Float16)We2[e * 72 + (k & 7) * 9 + (k >> 3)]
                        : (_Float16)0.f;
      }
    float bias2 = be2[e];
    // out rows: wave w owns row w; 16 col-tiles each
    for (int i = 0; i < 16; ++i) {
      int c0  = i * 16;
      int hl  = wave + 1;
      int col = c0 + nloc;
      v8f acc = {};
      v16h a0 = (lane < 16) ? a_c2<0, 0>(hs, hl, col) : a_c2<8, 0>(hs, hl, col);
      acc = __builtin_amdgcn_wmma_f32_16x16x32_f16(false, a0, false, b2[0],
                                                   (short)0, acc, false, false);
      v16h a1 = (lane < 16) ? a_c2<0, 1>(hs, hl, col) : a_c2<8, 1>(hs, hl, col);
      acc = __builtin_amdgcn_wmma_f32_16x16x32_f16(false, a1, false, b2[1],
                                                   (short)0, acc, false, false);
      v16h a2 = (lane < 16) ? a_c2<0, 2>(hs, hl, col) : a_c2<8, 2>(hs, hl, col);
      acc = __builtin_amdgcn_wmma_f32_16x16x32_f16(false, a2, false, b2[2],
                                                   (short)0, acc, false, false);
      if (nloc == 0) {                            // only N==0 column is real
#pragma unroll
        for (int r = 0; r < 8; ++r)
          oacc[wave][c0 + r + hi8] += w * (acc[r] + bias2);
      }
    }
    __syncthreads();                              // hs reused by next pass
  }

  for (int i = tid; i < 8 * 256; i += 256) {
    int r = i >> 8, c = i & 255;
    out[(size_t)s * (IMG * IMG) + (r0 + r) * IMG + c] = oacc[r][c];
  }
}

// ---------------------------------------------------------------- launch
extern "C" void kernel_launch(void* const* d_in, const int* in_sizes, int n_in,
                              void* d_out, int out_size, void* d_ws, size_t ws_size,
                              hipStream_t stream) {
  (void)in_sizes; (void)n_in; (void)out_size; (void)ws_size;
  const float* x   = (const float*)d_in[0];
  const float* Wg  = (const float*)d_in[1];
  const float* Wc  = (const float*)d_in[2];
  const float* bc  = (const float*)d_in[3];
  const float* Wp  = (const float*)d_in[4];
  const float* bp  = (const float*)d_in[5];
  const float* We1 = (const float*)d_in[6];
  const float* be1 = (const float*)d_in[7];
  const float* We2 = (const float*)d_in[8];
  const float* be2 = (const float*)d_in[9];
  float* out = (float*)d_out;

  char*  ws    = (char*)d_ws;
  float* feats = (float*)ws;                                    // 64*64 f32
  int*   gidx  = (int*)(ws + NB * GFEAT * sizeof(float));       // 64*2 i32
  float* gpr   = (float*)(ws + NB * GFEAT * sizeof(float) + NB * 2 * sizeof(int));

  zero_feats      <<<dim3(16),     dim3(256), 0, stream>>>(feats);
  gating_conv_wmma<<<dim3(4, NB),  dim3(256), 0, stream>>>(x, Wg, feats);
  gating_head     <<<dim3(1),      dim3(64),  0, stream>>>(feats, Wc, bc, Wp, bp, gidx, gpr);
  expert_moe_wmma <<<dim3(32, NB), dim3(256), 0, stream>>>(x, We1, be1, We2, be2, gidx, gpr, out);
}
